// SparseSelfAttention_8186207666183
// MI455X (gfx1250) — matrix operand order repaired and checked by reading, not compiled, not run on hardware
//
#include <hip/hip_runtime.h>

#define NN 50000
#define DD 512
#define HH 8
#define EE 3200000
#define DKK 64

typedef __attribute__((ext_vector_type(16))) __bf16 v16bf;
typedef __attribute__((ext_vector_type(8))) float v8f;

union BF16Frag { v16bf v; unsigned int u[8]; };

__device__ __forceinline__ unsigned short f2bf(float f) {
  unsigned int u = __float_as_uint(f);
  u += 0x7FFFu + ((u >> 16) & 1u);   // round-to-nearest-even
  return (unsigned short)(u >> 16);
}
__device__ __forceinline__ float bf2f(unsigned short h) {
  return __uint_as_float(((unsigned int)h) << 16);
}

// A fragment: 16x32 bf16 tile, row-major source [*, DD].
// Lane L: M = m0 + (L&15); lanes 16-31 offset K by +8.
// VGPR v holds K pair: (v&3)*2 + (v>>2)*16 (+8 for upper lanes).
__device__ __forceinline__ v16bf load_a_frag(const unsigned short* __restrict__ X,
                                             int m0, int kb, int lane) {
  BF16Frag r;
  const unsigned short* p = X + (size_t)(m0 + (lane & 15)) * DD + kb + (((lane >> 4) & 1) << 3);
#pragma unroll
  for (int v = 0; v < 8; ++v) {
    int k = ((v & 3) << 1) + ((v >> 2) << 4);
    r.u[v] = *(const unsigned int*)(p + k);   // two packed bf16
  }
  return r.v;
}

// B fragment: 32x16 bf16, B(k,j) = W[j, k] with W row-major [DD,DD] (i.e. B = W^T tile).
// Lane L: N = j0 + (L&15); lanes 16-31 offset K by +8; VGPR v holds K pair (v&3)*2 + (v>>2)*16.
__device__ __forceinline__ v16bf load_b_frag(const unsigned short* __restrict__ W,
                                             int j0, int kb, int lane) {
  BF16Frag r;
  const unsigned short* p = W + (size_t)(j0 + (lane & 15)) * DD + kb + (((lane >> 4) & 1) << 3);
#pragma unroll
  for (int v = 0; v < 8; ++v) r.u[v] = *(const unsigned int*)(p + ((v & 3) << 1) + ((v >> 2) << 4));
  return r.v;
}

// ---- f32 -> bf16 conversion of x and the four weight matrices ----
__global__ void convert_kernel(const float* __restrict__ x,
                               const float* __restrict__ Wq, const float* __restrict__ Wk,
                               const float* __restrict__ Wv, const float* __restrict__ Wo,
                               unsigned short* __restrict__ xb, unsigned short* __restrict__ wb) {
  size_t i = (size_t)blockIdx.x * blockDim.x + threadIdx.x;
  const size_t ND = (size_t)NN * DD;
  const size_t WD = (size_t)DD * DD;
  size_t total = ND + 4 * WD;
  if (i >= total) return;
  if (i < ND) { xb[i] = f2bf(x[i]); return; }
  size_t j = i - ND;
  int mIdx = (int)(j / WD);
  size_t o = j % WD;
  const float* src = (mIdx == 0) ? Wq : (mIdx == 1) ? Wk : (mIdx == 2) ? Wv : Wo;
  wb[(size_t)mIdx * WD + o] = f2bf(src[o]);
}

// ---- CSR row pointers from sorted row_index (lower_bound per row) ----
__global__ void row_ptr_kernel(const int* __restrict__ row_index, int* __restrict__ row_ptr) {
  int n = blockIdx.x * blockDim.x + threadIdx.x;
  if (n > NN) return;
  int lo = 0, hi = EE;
  while (lo < hi) {
    int mid = (lo + hi) >> 1;
    if (row_index[mid] < n) lo = mid + 1; else hi = mid;
  }
  row_ptr[n] = lo;
}

// ---- fused QKV projection GEMM: qkv[mat] = (x @ W^T + b) * scale, stored bf16 ----
__global__ __launch_bounds__(128) void qkv_gemm_kernel(
    const unsigned short* __restrict__ xb, const unsigned short* __restrict__ wb,
    const float* __restrict__ bq, const float* __restrict__ bk, const float* __restrict__ bv,
    unsigned short* __restrict__ qkv) {
  int lane = threadIdx.x & 31;
  int wave = threadIdx.x >> 5;
  int m0 = blockIdx.x << 4;
  int j0 = ((blockIdx.y << 2) + wave) << 4;
  int mat = blockIdx.z;
  const unsigned short* W = wb + (size_t)mat * DD * DD;
  const float* bias = (mat == 0) ? bq : ((mat == 1) ? bk : bv);
  float scale = (mat == 0) ? 0.125f : 1.0f;   // 1/sqrt(64) for Q
  v8f c = {};
#pragma unroll
  for (int kb = 0; kb < DD; kb += 32) {
    v16bf a = load_a_frag(xb, m0, kb, lane);
    v16bf b = load_b_frag(W, j0, kb, lane);
    c = __builtin_amdgcn_wmma_f32_16x16x32_bf16(false, a, false, b, (short)0, c, false, false);
  }
  int j = j0 + (lane & 15);
  int rofs = ((lane >> 4) & 1) << 3;
  float bj = bias[j];
  unsigned short* dst = qkv + (size_t)mat * NN * DD;
#pragma unroll
  for (int v = 0; v < 8; ++v)
    dst[(size_t)(m0 + rofs + v) * DD + j] = f2bf((c[v] + bj) * scale);
}

// ---- fused sparse attention, one block per (row, head), online softmax ----
__global__ __launch_bounds__(64) void sparse_attn_kernel(
    const unsigned short* __restrict__ qkv, const int* __restrict__ row_ptr,
    const int* __restrict__ col_index, const float* __restrict__ att_bias,
    unsigned short* __restrict__ yb) {
  __shared__ float qs[DKK];
  __shared__ float red[64];
  __shared__ float pbuf[64];
  __shared__ int colbuf[64];
  int n = blockIdx.x;
  int h = blockIdx.y;
  int t = threadIdx.x;
  const unsigned short* qb = qkv;
  const unsigned short* kmat = qkv + (size_t)NN * DD;
  const unsigned short* vmat = qkv + (size_t)2 * NN * DD;
  qs[t] = bf2f(qb[(size_t)n * DD + h * DKK + t]);
  __syncthreads();
  int start = row_ptr[n], end = row_ptr[n + 1];
  float m = -3.0e38f, s = 0.0f, acc = 0.0f;
  for (int e0 = start; e0 < end; e0 += 64) {
    int cnt = min(64, end - e0);
    float l = -3.0e38f;
    if (t < cnt) {
      int col = col_index[e0 + t];
      colbuf[t] = col;
      const unsigned short* kr = kmat + (size_t)col * DD + h * DKK;
      float d = 0.0f;
#pragma unroll 8
      for (int i = 0; i < DKK; ++i) d += qs[i] * bf2f(kr[i]);
      l = d + att_bias[(size_t)h * EE + e0 + t];
    }
    red[t] = l;
    __syncthreads();
#pragma unroll
    for (int off = 32; off > 0; off >>= 1) {
      if (t < off) red[t] = fmaxf(red[t], red[t + off]);
      __syncthreads();
    }
    float cm = red[0];
    __syncthreads();
    float mnew = fmaxf(m, cm);
    float rescale = __expf(m - mnew);   // 0 on first chunk
    float p = (t < cnt) ? __expf(l - mnew) : 0.0f;
    pbuf[t] = p;
    red[t] = p;
    __syncthreads();
#pragma unroll
    for (int off = 32; off > 0; off >>= 1) {
      if (t < off) red[t] += red[t + off];
      __syncthreads();
    }
    s = s * rescale + red[0];
    acc *= rescale;
    for (int jj = 0; jj < cnt; ++jj)
      acc += pbuf[jj] * bf2f(vmat[(size_t)colbuf[jj] * DD + h * DKK + t]);
    m = mnew;
    __syncthreads();
  }
  float outv = (end > start) ? (acc / s) : 0.0f;
  yb[(size_t)n * DD + h * DKK + t] = f2bf(outv);
}

// ---- output GEMM: out = y @ Wo^T + bo, f32 result ----
__global__ __launch_bounds__(128) void out_gemm_kernel(
    const unsigned short* __restrict__ yb, const unsigned short* __restrict__ wo,
    const float* __restrict__ bo, float* __restrict__ out) {
  int lane = threadIdx.x & 31;
  int wave = threadIdx.x >> 5;
  int m0 = blockIdx.x << 4;
  int j0 = ((blockIdx.y << 2) + wave) << 4;
  v8f c = {};
#pragma unroll
  for (int kb = 0; kb < DD; kb += 32) {
    v16bf a = load_a_frag(yb, m0, kb, lane);
    v16bf b = load_b_frag(wo, j0, kb, lane);
    c = __builtin_amdgcn_wmma_f32_16x16x32_bf16(false, a, false, b, (short)0, c, false, false);
  }
  int j = j0 + (lane & 15);
  int rofs = ((lane >> 4) & 1) << 3;
  float bj = bo[j];
#pragma unroll
  for (int v = 0; v < 8; ++v)
    out[(size_t)(m0 + rofs + v) * DD + j] = c[v] + bj;
}

extern "C" void kernel_launch(void* const* d_in, const int* in_sizes, int n_in,
                              void* d_out, int out_size, void* d_ws, size_t ws_size,
                              hipStream_t stream) {
  const float* x        = (const float*)d_in[0];
  const int* row_index  = (const int*)d_in[1];
  const int* col_index  = (const int*)d_in[2];
  const float* att_bias = (const float*)d_in[3];
  const float* Wq = (const float*)d_in[4];
  const float* bq = (const float*)d_in[5];
  const float* Wk = (const float*)d_in[6];
  const float* bk = (const float*)d_in[7];
  const float* Wv = (const float*)d_in[8];
  const float* bv = (const float*)d_in[9];
  const float* Wo = (const float*)d_in[10];
  const float* bo = (const float*)d_in[11];
  float* out = (float*)d_out;

  // workspace layout (bf16 unless noted): xb | wb(q,k,v,o) | qkv(3) | yb | row_ptr
  unsigned short* xb  = (unsigned short*)d_ws;
  unsigned short* wb  = xb + (size_t)NN * DD;
  unsigned short* qkv = wb + 4 * (size_t)DD * DD;
  unsigned short* yb  = qkv + 3 * (size_t)NN * DD;
  int* row_ptr        = (int*)(yb + (size_t)NN * DD);

  {
    size_t total = (size_t)NN * DD + 4 * (size_t)DD * DD;
    int blocks = (int)((total + 255) / 256);
    convert_kernel<<<blocks, 256, 0, stream>>>(x, Wq, Wk, Wv, Wo, xb, wb);
  }
  row_ptr_kernel<<<(NN + 1 + 255) / 256, 256, 0, stream>>>(row_index, row_ptr);
  qkv_gemm_kernel<<<dim3(NN / 16, DD / 64, 3), 128, 0, stream>>>(xb, wb, bq, bk, bv, qkv);
  sparse_attn_kernel<<<dim3(NN, HH), 64, 0, stream>>>(qkv, row_ptr, col_index, att_bias, yb);
  out_gemm_kernel<<<dim3(NN / 16, DD / 64), 128, 0, stream>>>(yb, wb + 3 * (size_t)DD * DD, bo, out);
}
